// RAFT_3985729651364
// MI455X (gfx1250) — matrix operand order repaired and checked by reading, not compiled, not run on hardware
//
#include <hip/hip_runtime.h>
#include <math.h>

// ---------------------------------------------------------------------------
// RAFT update step for MI455X (gfx1250, wave32).
// GEMM-shaped work (corr pyramid, 1x1 convs, im2col'd 3x3 convs, MLPs) runs on
// V_WMMA_F32_16X16X32_F16, f32 accumulate. A operand is M-major and feeds
// fragments DIRECTLY from global (L0/L2-cached, two b128 per frag); only the
// B operand (K-major activations) is transposed through LDS. 32x32 output per
// wave = 4 WMMA per k-step. fp16 corr pyramid (~89MB) fits the 192MB L2.
// ---------------------------------------------------------------------------

typedef _Float16 half_t;
typedef __attribute__((ext_vector_type(16))) _Float16 v16h;
typedef __attribute__((ext_vector_type(8)))  _Float16 v8h;
typedef __attribute__((ext_vector_type(2)))  _Float16 v2h;
typedef __attribute__((ext_vector_type(8)))  float    v8f;

#define NT 8192          // tokens = B*H*W
#define HW 4096          // H*W

// =============================== WMMA GEMM =================================
// C[M x N] = act(alpha * A x B + bias), batched via grid.z.
// A M-major: A[m*ldk + k]  (rows padded to multiple of 128, k to mult of 32)
// B K-major: B[k*ldb + n]
// BM=128, BN=64, BK=32; 8 waves per block, 32x32 tile per wave (4 WMMA).
#define BM 128
#define BN 64
#define BK 32

union FragH { v16h v; v8h h[2]; };

template<int ACT>
__global__ __launch_bounds__(256) void gemm_kernel(
    const half_t* __restrict__ A, const half_t* __restrict__ B,
    const float* __restrict__ bias, float* __restrict__ Cf, half_t* __restrict__ Ch,
    int M, int N, int K, int ldk, int ldb, int ldc,
    long strideA, long strideB, long strideC, float alpha)
{
  __shared__ half_t Bs[BN][BK + 8];   // [n][k], transposed staging, padded row

  const int bz = blockIdx.z;
  A += (long)bz * strideA;
  B += (long)bz * strideB;

  const int m0 = blockIdx.y * BM;
  const int n0 = blockIdx.x * BN;
  const int t  = threadIdx.x;
  const int lane = t & 31;
  const int wave = t >> 5;                 // 0..7
  const int wm = (wave & 3) * 32;          // wave tile row offset in block
  const int wn = (wave >> 2) * 32;         // wave tile col offset in block
  const int row = lane & 15;
  const int hi  = lane >> 4;               // 0: lanes 0-15, 1: lanes 16-31
  const int kbA = hi * 8;                  // A frag K runs: kbA..+7, kbA+16..+23
  const int kbB = hi * 16;                 // unused: B frags load the same kbA runs as A

  // A fragment source rows (direct global, M-major)
  const half_t* Ar0 = A + (long)(m0 + wm + row) * ldk;
  const half_t* Ar1 = Ar0 + 16L * ldk;

  // B staging assignment: 8 halves per thread = (4 n) x (2 k)
  const int k2 = (t & 15) * 2;
  const int nb = (t >> 4) * 4;

  v8f acc[2][2] = {};

  for (int k0 = 0; k0 < K; k0 += BK) {
    // ---- A fragments: direct 16B global loads (issued early, overlap B fill)
    FragH af0, af1;
    af0.h[0] = *(const v8h*)(Ar0 + k0 + kbA);
    af0.h[1] = *(const v8h*)(Ar0 + k0 + kbA + 16);
    af1.h[0] = *(const v8h*)(Ar1 + k0 + kbA);
    af1.h[1] = *(const v8h*)(Ar1 + k0 + kbA + 16);

    // ---- stage B block [BK x BN] -> Bs[n][k], paired-k b32 stores ----
    {
      const half_t* gb = B + (long)(k0 + k2) * ldb + n0 + nb;
      union { uint2 u; half_t h[4]; } r0, r1;
      r0.u = *(const uint2*)gb;
      r1.u = *(const uint2*)(gb + ldb);
      #pragma unroll
      for (int i = 0; i < 4; ++i)
        *(v2h*)&Bs[nb + i][k2] = (v2h){r0.h[i], r1.h[i]};
    }
    // prefetch next k-step (global_prefetch_b8)
    if (k0 + BK < K) {
      __builtin_prefetch((const void*)(Ar0 + k0 + BK + kbA), 0, 3);
      __builtin_prefetch((const void*)(B + (long)(k0 + BK + k2) * ldb + n0 + nb), 0, 3);
    }
    __syncthreads();

    // ---- B fragments from LDS (vector b128 loads) ----
    FragH bf0, bf1;
    bf0.h[0] = *(const v8h*)&Bs[wn + row][kbA];
    bf0.h[1] = *(const v8h*)&Bs[wn + row][kbA + 16];
    bf1.h[0] = *(const v8h*)&Bs[wn + 16 + row][kbA];
    bf1.h[1] = *(const v8h*)&Bs[wn + 16 + row][kbA + 16];

    acc[0][0] = __builtin_amdgcn_wmma_f32_16x16x32_f16(
        false, af0.v, false, bf0.v, (short)0, acc[0][0], false, false);
    acc[0][1] = __builtin_amdgcn_wmma_f32_16x16x32_f16(
        false, af0.v, false, bf1.v, (short)0, acc[0][1], false, false);
    acc[1][0] = __builtin_amdgcn_wmma_f32_16x16x32_f16(
        false, af1.v, false, bf0.v, (short)0, acc[1][0], false, false);
    acc[1][1] = __builtin_amdgcn_wmma_f32_16x16x32_f16(
        false, af1.v, false, bf1.v, (short)0, acc[1][1], false, false);
    __syncthreads();
  }

  // ---- epilogue: C VGPR r -> row m = r + hi*8, col n = lane&15 ----
  #pragma unroll
  for (int ti = 0; ti < 2; ++ti) {
    #pragma unroll
    for (int tj = 0; tj < 2; ++tj) {
      const int n = n0 + wn + tj * 16 + row;
      #pragma unroll
      for (int r = 0; r < 8; ++r) {
        int mm = m0 + wm + ti * 16 + r + hi * 8;
        float v = acc[ti][tj][r] * alpha;
        if (bias) v += bias[mm < M ? mm : 0];
        if (ACT == 1) v = fmaxf(v, 0.f);
        else if (ACT == 2) v = 0.5f * v * (1.f + erff(v * 0.70710678118f));
        if (mm < M) {
          long off = (long)bz * strideC + (long)mm * ldc + n;
          if (Cf) Cf[off] = v;
          if (Ch) Ch[off] = (half_t)v;
        }
      }
    }
  }
}

static void launch_gemm(int act, const half_t* A, const half_t* B, const float* bias,
                        float* Cf, half_t* Ch, int M, int N, int K,
                        int ldk, int ldb, int ldc,
                        long sA, long sB, long sC, int batch, float alpha,
                        hipStream_t s)
{
  dim3 grid((unsigned)(N / BN), (unsigned)((M + BM - 1) / BM), (unsigned)batch);
  dim3 block(256);
  switch (act) {
    case 0: gemm_kernel<0><<<grid, block, 0, s>>>(A,B,bias,Cf,Ch,M,N,K,ldk,ldb,ldc,sA,sB,sC,alpha); break;
    case 1: gemm_kernel<1><<<grid, block, 0, s>>>(A,B,bias,Cf,Ch,M,N,K,ldk,ldb,ldc,sA,sB,sC,alpha); break;
    default: gemm_kernel<2><<<grid, block, 0, s>>>(A,B,bias,Cf,Ch,M,N,K,ldk,ldb,ldc,sA,sB,sC,alpha); break;
  }
}

// ======================= layout / elementwise kernels ======================
__global__ void k_f32_to_f16(const float* __restrict__ in, half_t* __restrict__ out, long n) {
  long i = (long)blockIdx.x * 256 + threadIdx.x;
  if (i < n) out[i] = (half_t)in[i];
}

__global__ void k_fill_zero_h(half_t* p, long n) {
  long i = (long)blockIdx.x * 256 + threadIdx.x;
  if (i < n) p[i] = (half_t)0.f;
}

__global__ void k_avgpool2(const float* __restrict__ in, float* __restrict__ out,
                           int C, int h, int w) {
  int oh = h >> 1, ow = w >> 1;
  long n = (long)C * oh * ow;
  long i = (long)blockIdx.x * 256 + threadIdx.x;
  if (i >= n) return;
  int x = (int)(i % ow);
  int y = (int)((i / ow) % oh);
  int c = (int)(i / ((long)ow * oh));
  const float* p = in + ((long)c * h + 2 * y) * w + 2 * x;
  out[i] = 0.25f * (p[0] + p[1] + p[w] + p[w + 1]);
}

// [B][C][HW] f32 -> [C][B*HW] f32
__global__ void k_nchw_to_cn(const float* __restrict__ in, float* __restrict__ out, int C) {
  long n = (long)C * NT;
  long i = (long)blockIdx.x * 256 + threadIdx.x;
  if (i >= n) return;
  int c = (int)(i / NT), r = (int)(i % NT);
  int b = r / HW, p = r % HW;
  out[i] = in[((long)b * C + c) * HW + p];
}

// [C][B*HW] f32 -> [B][C][HW] f32
__global__ void k_cn_to_nchw(const float* __restrict__ in, float* __restrict__ out, int C) {
  long n = 2L * C * HW;
  long i = (long)blockIdx.x * 256 + threadIdx.x;
  if (i >= n) return;
  int b = (int)(i / ((long)C * HW));
  int rem = (int)(i % ((long)C * HW));
  int c = rem / HW, p = rem % HW;
  out[i] = in[(long)c * NT + b * HW + p];
}

// fmap1 [B][256][4096] f32 -> f1t [B][4096][256] f16 (M-major A for corr GEMM)
__global__ void k_fmap1_tr(const float* __restrict__ in, half_t* __restrict__ out) {
  long n = 2L * 4096 * 256;
  long i = (long)blockIdx.x * 256 + threadIdx.x;
  if (i >= n) return;
  int d = (int)(i & 255);
  int nn = (int)((i >> 8) & 4095);
  int b = (int)(i >> 20);
  out[i] = (half_t)in[((long)b * 256 + d) * 4096 + nn];
}

// f32 [M][K] row-major -> f16 [Mpad][Kpad] M-major, zero padded
__global__ void k_pad_convert(const float* __restrict__ in, half_t* __restrict__ out,
                              int M, int K, int Mpad, int Kpad) {
  long n = (long)Mpad * Kpad;
  long i = (long)blockIdx.x * 256 + threadIdx.x;
  if (i >= n) return;
  int k = (int)(i % Kpad);
  int m = (int)(i / Kpad);
  float v = (m < M && k < K) ? in[(long)m * K + k] : 0.f;
  out[i] = (half_t)v;
}

// f32 [K][M] (K-major source) -> f16 [Mpad][Kpad] M-major, zero padded
__global__ void k_pad_convert_t(const float* __restrict__ in, half_t* __restrict__ out,
                                int M, int K, int Mpad, int Kpad) {
  long n = (long)Mpad * Kpad;
  long i = (long)blockIdx.x * 256 + threadIdx.x;
  if (i >= n) return;
  int k = (int)(i % Kpad);
  int m = (int)(i / Kpad);
  float v = (m < M && k < K) ? in[(long)k * M + m] : 0.f;
  out[i] = (half_t)v;
}

// X f16 [C][B*HW] -> col f16 [C*9][B*HW], 3x3 pad 1
__global__ void k_im2col3(const half_t* __restrict__ X, half_t* __restrict__ col, int C) {
  long n = (long)C * 9 * NT;
  long i = (long)blockIdx.x * 256 + threadIdx.x;
  if (i >= n) return;
  int tok = (int)(i % NT);
  long r = i / NT;
  int t = (int)(r % 9);
  int ic = (int)(r / 9);
  int b = tok >> 12, p = tok & 4095, y = p >> 6, x = p & 63;
  int iy = y + t / 3 - 1, ix = x + t % 3 - 1;
  half_t v = (half_t)0.f;
  if (iy >= 0 && iy < 64 && ix >= 0 && ix < 64)
    v = X[(long)ic * NT + (b << 12) + iy * 64 + ix];
  col[i] = v;
}

// correlation lookup: pyr f16, center = coords0+flow, 4 levels x 81 taps
__global__ void k_corr_lookup(const half_t* __restrict__ pyr, const float* __restrict__ flow,
                              half_t* __restrict__ out) {
  int i = blockIdx.x * 256 + threadIdx.x;
  if (i >= 4 * NT) return;
  int lvl = i >> 13;
  int tok = i & (NT - 1);
  int b = tok >> 12, p = tok & 4095, y = p >> 6, x = p & 63;
  float fx = flow[(long)(b * 2 + 0) * HW + p] + (float)x;
  float fy = flow[(long)(b * 2 + 1) * HW + p] + (float)y;
  int dim = 64 >> lvl;
  float inv = 1.f / (float)(1 << lvl);
  float cx = fx * inv, cy = fy * inv;
  const long b1 = 8192L * 4096;
  const long b2 = b1 + 8192L * 1024;
  const long b3 = b2 + 8192L * 256;
  long base = (lvl == 0) ? 0 : (lvl == 1) ? b1 : (lvl == 2) ? b2 : b3;
  const half_t* plane = pyr + base + (long)tok * dim * dim;
  for (int a = 0; a < 9; ++a) {
    float xs = cx + (float)(a - 4);       // delta[...,0] = dy grid -> x coord (RAFT quirk)
    float x0 = floorf(xs);
    int ix = (int)x0;
    float wx = xs - x0;
    for (int c = 0; c < 9; ++c) {
      float ys = cy + (float)(c - 4);
      float y0 = floorf(ys);
      int iy = (int)y0;
      float wy = ys - y0;
      float s = 0.f;
      #pragma unroll
      for (int ty = 0; ty < 2; ++ty) {
        #pragma unroll
        for (int tx = 0; tx < 2; ++tx) {
          int yi = iy + ty, xi = ix + tx;
          if (xi >= 0 && xi < dim && yi >= 0 && yi < dim) {
            float wgt = (ty ? wy : 1.f - wy) * (tx ? wx : 1.f - wx);
            s += wgt * (float)plane[yi * dim + xi];
          }
        }
      }
      out[(long)(lvl * 81 + a * 9 + c) * NT + tok] = (half_t)s;
    }
  }
}

// depthwise 7x7 on [384][B*HW] f32
__global__ void k_dwconv7(const float* __restrict__ x, const float* __restrict__ w,
                          const float* __restrict__ bias, float* __restrict__ y) {
  long n = 384L * NT;
  long i = (long)blockIdx.x * 256 + threadIdx.x;
  if (i >= n) return;
  int tok = (int)(i % NT);
  int c = (int)(i / NT);
  int b = tok >> 12, p = tok & 4095, yy = p >> 6, xx = p & 63;
  float acc = bias[c];
  const float* xp = x + (long)c * NT + (b << 12);
  const float* wp = w + c * 49;
  for (int ky = 0; ky < 7; ++ky) {
    int iy = yy + ky - 3;
    if (iy < 0 || iy >= 64) continue;
    for (int kx = 0; kx < 7; ++kx) {
      int ix = xx + kx - 3;
      if (ix < 0 || ix >= 64) continue;
      acc += xp[iy * 64 + ix] * wp[ky * 7 + kx];
    }
  }
  y[i] = acc;
}

// 7x7 conv, 2->128, pad 3, relu; reads flow NCHW directly, writes f16 CN
__global__ void k_conv7_flow(const float* __restrict__ flow, const float* __restrict__ w,
                             const float* __restrict__ bias, half_t* __restrict__ out) {
  long n = 128L * NT;
  long i = (long)blockIdx.x * 256 + threadIdx.x;
  if (i >= n) return;
  int tok = (int)(i % NT);
  int oc = (int)(i / NT);
  int b = tok >> 12, p = tok & 4095, yy = p >> 6, xx = p & 63;
  float acc = bias[oc];
  for (int ic = 0; ic < 2; ++ic) {
    const float* xp = flow + (long)(b * 2 + ic) * HW;
    const float* wp = w + ((long)oc * 2 + ic) * 49;
    for (int ky = 0; ky < 7; ++ky) {
      int iy = yy + ky - 3;
      if (iy < 0 || iy >= 64) continue;
      for (int kx = 0; kx < 7; ++kx) {
        int ix = xx + kx - 3;
        if (ix < 0 || ix >= 64) continue;
        acc += xp[iy * 64 + ix] * wp[ky * 7 + kx];
      }
    }
  }
  out[i] = (half_t)fmaxf(acc, 0.f);
}

// channel LayerNorm over 384 per token, affine, write f16
__global__ void k_layernorm(const float* __restrict__ x, const float* __restrict__ g,
                            const float* __restrict__ bta, half_t* __restrict__ out) {
  int tok = blockIdx.x * 256 + threadIdx.x;
  if (tok >= NT) return;
  float mu = 0.f;
  for (int c = 0; c < 384; ++c) mu += x[(long)c * NT + tok];
  mu *= (1.f / 384.f);
  float var = 0.f;
  for (int c = 0; c < 384; ++c) {
    float d = x[(long)c * NT + tok] - mu;
    var += d * d;
  }
  var *= (1.f / 384.f);
  float inv = rsqrtf(var + 1e-6f);
  for (int c = 0; c < 384; ++c) {
    float v = (x[(long)c * NT + tok] - mu) * inv * g[c] + bta[c];
    out[(long)c * NT + tok] = (half_t)v;
  }
}

// comboh = f16(xin + gamma[c] * p2out)   (ConvNeXt residual, pre-fw)
__global__ void k_residual(const float* __restrict__ xin, const float* __restrict__ p2,
                           const float* __restrict__ gamma, half_t* __restrict__ out) {
  long n = 384L * NT;
  long i = (long)blockIdx.x * 256 + threadIdx.x;
  if (i >= n) return;
  int c = (int)(i / NT);
  out[i] = (half_t)(xin[i] + gamma[c] * p2[i]);
}

// ================================ host =====================================
#define GRID1(n) dim3((unsigned)(((n) + 255) / 256)), dim3(256), 0, stream

extern "C" void kernel_launch(void* const* d_in, const int* in_sizes, int n_in,
                              void* d_out, int out_size, void* d_ws, size_t ws_size,
                              hipStream_t stream) {
  const float *fmap1, *fmap2, *net, *inp, *flow;
  const float *c1w, *c1b, *c2w, *c2b, *f1w, *f1b, *f2w, *f2b, *cww, *cwb;
  const float *dww[2], *dwb[2], *lnw[2], *lnb[2], *p1w[2], *p1b[2],
              *p2w[2], *p2b[2], *gmm[2], *fww[2], *fwb[2];
  auto F = [&](int i) -> const float* { return (const float*)d_in[i]; };

  if (in_sizes[0] == 16384) {
    // jax tree-leaves order (all dict keys sorted)
    flow = F(0); fmap1 = F(1); fmap2 = F(2); inp = F(3); net = F(4);
    for (int b = 0; b < 2; ++b) {
      int o = 5 + 11 * b;
      dwb[b] = F(o + 0); dww[b] = F(o + 1); fwb[b] = F(o + 2); fww[b] = F(o + 3);
      gmm[b] = F(o + 4); lnb[b] = F(o + 5); lnw[b] = F(o + 6); p1b[b] = F(o + 7);
      p1w[b] = F(o + 8); p2b[b] = F(o + 9); p2w[b] = F(o + 10);
    }
    int o = 27;
    c1b = F(o + 0); c1w = F(o + 1); c2b = F(o + 2); c2w = F(o + 3); cwb = F(o + 4);
    cww = F(o + 5); f1b = F(o + 6); f1w = F(o + 7); f2b = F(o + 8); f2w = F(o + 9);
  } else {
    // dict insertion order
    fmap1 = F(0); fmap2 = F(1); net = F(2); inp = F(3); flow = F(4);
    int o = 5;
    c1w = F(o + 0); c1b = F(o + 1); c2w = F(o + 2); c2b = F(o + 3); f1w = F(o + 4);
    f1b = F(o + 5); f2w = F(o + 6); f2b = F(o + 7); cww = F(o + 8); cwb = F(o + 9);
    for (int b = 0; b < 2; ++b) {
      int q = 15 + 11 * b;
      dww[b] = F(q + 0); dwb[b] = F(q + 1); lnw[b] = F(q + 2); lnb[b] = F(q + 3);
      p1w[b] = F(q + 4); p1b[b] = F(q + 5); p2w[b] = F(q + 6); p2b[b] = F(q + 7);
      gmm[b] = F(q + 8); fww[b] = F(q + 9); fwb[b] = F(q + 10);
    }
  }

  // ---- workspace bump allocator (256B aligned) ----
  char* wsb = (char*)d_ws;
  size_t off = 0;
  auto alloc = [&](size_t bytes) -> void* {
    void* r = wsb + off;
    off += (bytes + 255) & ~(size_t)255;
    return r;
  };
  half_t* f1t   = (half_t*)alloc(2L * 4096 * 256 * 2);    // fmap1 transposed, M-major
  float*  f2l1  = (float*) alloc(2L * 256 * 1024 * 4);
  float*  f2l2  = (float*) alloc(2L * 256 * 256 * 4);
  float*  f2l3  = (float*) alloc(2L * 256 * 64 * 4);
  half_t* f2h0  = (half_t*)alloc(2L * 256 * 4096 * 2);
  half_t* f2h1  = (half_t*)alloc(2L * 256 * 1024 * 2);
  half_t* f2h2  = (half_t*)alloc(2L * 256 * 256 * 2);
  half_t* f2h3  = (half_t*)alloc(2L * 256 * 64 * 2);
  half_t* pyr   = (half_t*)alloc(8192L * 5440 * 2);       // fp16 corr pyramid (~89MB, fits L2)
  half_t* corrh = (half_t*)alloc(352L * NT * 2);          // 324 chans padded to 352
  half_t* A1    = (half_t*)alloc(256L * 352 * 2);         // [Mpad][Kpad] M-major
  half_t* A2    = (half_t*)alloc(256L * 2304 * 2);        // 192 -> 256 rows
  half_t* A3    = (half_t*)alloc(128L * 1152 * 2);        // 64 -> 128 rows
  half_t* A4    = (half_t*)alloc(128L * 2304 * 2);        // 126 -> 128 rows
  half_t *Ap1[2], *Ap2[2], *Afw[2];
  for (int b = 0; b < 2; ++b) {
    Ap1[b] = (half_t*)alloc(512L * 384 * 2);
    Ap2[b] = (half_t*)alloc(384L * 512 * 2);
    Afw[b] = (half_t*)alloc(128L * 384 * 2);
  }
  half_t* cor1h = (half_t*)alloc(256L * NT * 2);
  half_t* colb  = (half_t*)alloc(2304L * NT * 2);
  half_t* catA  = (half_t*)alloc(256L * NT * 2);          // [cor2(192); flo2(64)]
  half_t* flo1h = (half_t*)alloc(128L * NT * 2);
  float*  xin   = (float*) alloc(384L * NT * 4);          // [net(128); inp(128); mf(126); flow(2)]
  float*  dwout = (float*) alloc(384L * NT * 4);
  half_t* lnh   = (half_t*)alloc(384L * NT * 2);
  half_t* p1h   = (half_t*)alloc(512L * NT * 2);
  float*  p2f   = (float*) alloc(384L * NT * 4);
  half_t* cmb   = (half_t*)alloc(384L * NT * 2);
  (void)ws_size; (void)n_in; (void)out_size;

  // ---- feature pyramid (avgpool) + fp16 conversion ----
  k_fmap1_tr<<<GRID1(2L * 4096 * 256)>>>(fmap1, f1t);
  k_avgpool2<<<GRID1(2L * 256 * 1024)>>>(fmap2, f2l1, 512, 64, 64);
  k_avgpool2<<<GRID1(2L * 256 * 256)>>>(f2l1, f2l2, 512, 32, 32);
  k_avgpool2<<<GRID1(2L * 256 * 64)>>>(f2l2, f2l3, 512, 16, 16);
  k_f32_to_f16<<<GRID1(2L * 256 * 4096)>>>(fmap2, f2h0, 2L * 256 * 4096);
  k_f32_to_f16<<<GRID1(2L * 256 * 1024)>>>(f2l1, f2h1, 2L * 256 * 1024);
  k_f32_to_f16<<<GRID1(2L * 256 * 256)>>>(f2l2, f2h2, 2L * 256 * 256);
  k_f32_to_f16<<<GRID1(2L * 256 * 64)>>>(f2l3, f2h3, 2L * 256 * 64);

  // ---- weight prep (M-major f16, zero padded) ----
  k_pad_convert<<<GRID1(256L * 352)>>>(c1w, A1, 256, 324, 256, 352);
  k_pad_convert<<<GRID1(256L * 2304)>>>(c2w, A2, 192, 2304, 256, 2304);
  k_pad_convert<<<GRID1(128L * 1152)>>>(f2w, A3, 64, 1152, 128, 1152);
  k_pad_convert<<<GRID1(128L * 2304)>>>(cww, A4, 126, 2304, 128, 2304);
  for (int b = 0; b < 2; ++b) {
    k_pad_convert_t<<<GRID1(512L * 384)>>>(p1w[b], Ap1[b], 512, 384, 512, 384); // in [384][512]
    k_pad_convert_t<<<GRID1(384L * 512)>>>(p2w[b], Ap2[b], 384, 512, 384, 512); // in [512][384]
    k_pad_convert<<<GRID1(128L * 384)>>>(fww[b], Afw[b], 128, 384, 128, 384);
  }

  // ---- correlation pyramid GEMMs: C = (1/16) f1t x f2, per batch (grid.z=2) ----
  half_t* pyr0 = pyr;
  half_t* pyr1 = pyr0 + 8192L * 4096;
  half_t* pyr2 = pyr1 + 8192L * 1024;
  half_t* pyr3 = pyr2 + 8192L * 256;
  launch_gemm(0, f1t, f2h0, nullptr, nullptr, pyr0, 4096, 4096, 256,
              256, 4096, 4096, 4096L * 256, 256L * 4096, 4096L * 4096, 2, 0.0625f, stream);
  launch_gemm(0, f1t, f2h1, nullptr, nullptr, pyr1, 4096, 1024, 256,
              256, 1024, 1024, 4096L * 256, 256L * 1024, 4096L * 1024, 2, 0.0625f, stream);
  launch_gemm(0, f1t, f2h2, nullptr, nullptr, pyr2, 4096, 256, 256,
              256, 256, 256, 4096L * 256, 256L * 256, 4096L * 256, 2, 0.0625f, stream);
  launch_gemm(0, f1t, f2h3, nullptr, nullptr, pyr3, 4096, 64, 256,
              256, 64, 64, 4096L * 256, 256L * 64, 4096L * 64, 2, 0.0625f, stream);

  // ---- lookup -> corrh [352][NT] f16 (rows 324..351 zero) ----
  k_fill_zero_h<<<GRID1(28L * NT)>>>(corrh + 324L * NT, 28L * NT);
  k_corr_lookup<<<GRID1(4L * NT)>>>(pyr, flow, corrh);

  // ---- assemble xin constant rows: net | inp | (mf later) | flow ----
  k_nchw_to_cn<<<GRID1(128L * NT)>>>(net, xin, 128);
  k_nchw_to_cn<<<GRID1(128L * NT)>>>(inp, xin + 128L * NT, 128);
  k_nchw_to_cn<<<GRID1(2L * NT)>>>(flow, xin + 382L * NT, 2);

  // ---- motion encoder ----
  launch_gemm(1, A1, corrh, c1b, nullptr, cor1h, 256, NT, 352,
              352, NT, NT, 0, 0, 0, 1, 1.f, stream);                   // c1 + relu
  k_im2col3<<<GRID1(256L * 9 * NT)>>>(cor1h, colb, 256);
  launch_gemm(1, A2, colb, c2b, nullptr, catA, 192, NT, 2304,
              2304, NT, NT, 0, 0, 0, 1, 1.f, stream);                  // c2 + relu -> catA[0:192)
  k_conv7_flow<<<GRID1(128L * NT)>>>(flow, f1w, f1b, flo1h);           // f1 + relu
  k_im2col3<<<GRID1(128L * 9 * NT)>>>(flo1h, colb, 128);
  launch_gemm(1, A3, colb, f2b, nullptr, catA + 192L * NT, 64, NT, 1152,
              1152, NT, NT, 0, 0, 0, 1, 1.f, stream);                  // f2 + relu -> catA[192:256)
  k_im2col3<<<GRID1(256L * 9 * NT)>>>(catA, colb, 256);
  launch_gemm(1, A4, colb, cwb, xin + 256L * NT, nullptr, 126, NT, 2304,
              2304, NT, NT, 0, 0, 0, 1, 1.f, stream);                  // cw + relu -> xin rows 256..381

  // ---- 2 ConvNeXt blocks; fw writes new net into xin rows 0..127 ----
  for (int bl = 0; bl < 2; ++bl) {
    k_dwconv7<<<GRID1(384L * NT)>>>(xin, dww[bl], dwb[bl], dwout);
    k_layernorm<<<GRID1((long)NT)>>>(dwout, lnw[bl], lnb[bl], lnh);
    launch_gemm(2, Ap1[bl], lnh, p1b[bl], nullptr, p1h, 512, NT, 384,
                384, NT, NT, 0, 0, 0, 1, 1.f, stream);                 // p1 + exact GELU
    launch_gemm(0, Ap2[bl], p1h, p2b[bl], p2f, nullptr, 384, NT, 512,
                512, NT, NT, 0, 0, 0, 1, 1.f, stream);                 // p2
    k_residual<<<GRID1(384L * NT)>>>(xin, p2f, gmm[bl], cmb);
    launch_gemm(0, Afw[bl], cmb, fwb[bl], xin, nullptr, 128, NT, 384,
                384, NT, NT, 0, 0, 0, 1, 1.f, stream);                 // fw -> new net
  }

  // ---- final output: xin rows 0..127 (CN) -> NCHW f32 ----
  k_cn_to_nchw<<<GRID1(2L * 128 * HW)>>>(xin, (float*)d_out, 128);
}
